// LinkPredictor_5446018531553
// MI455X (gfx1250) — matrix operand order repaired and checked by reading, not compiled, not run on hardware
//
#include <hip/hip_runtime.h>

// ---------------------------------------------------------------------------
// LinkPredictor for MI455X (gfx1250, wave32, WMMA bf16 16x16x32)
//
// Roofline: A [16384^2] fp32 is read twice (2.15 GB) -> ~92us HBM floor at
// 23.3 TB/s; total math is only ~104 GFLOP, trivially hidden by bf16 WMMA.
// So the aggregation kernel is built as a bandwidth machine:
//   - NT fp32 loads of A into registers, cvt_pk_bf16 into padded LDS
//   - PT panel streamed global->LDS via gfx1250 async-LDS copies (ASYNCcnt)
//   - double-buffered LDS so next-chunk loads overlap current-chunk WMMA
// ---------------------------------------------------------------------------

typedef float        f32x4  __attribute__((ext_vector_type(4)));
typedef unsigned int u32x4t __attribute__((ext_vector_type(4)));
typedef unsigned int u32x2t __attribute__((ext_vector_type(2)));
typedef __bf16       v16bf  __attribute__((ext_vector_type(16)));
typedef float        v8f    __attribute__((ext_vector_type(8)));

union FragAB { v16bf v; u32x4t q[2]; };

static constexpr int N_NODES = 16384;
static constexpr int H_DIM   = 64;
static constexpr int E_EDGES = 524288;

__device__ __forceinline__ v8f wmma_bf16(const FragAB& a, const FragAB& b, v8f c) {
  return __builtin_amdgcn_wmma_f32_16x16x32_bf16(
      false, a.v, false, b.v, (short)0, c, false, false);
}

// ---- gfx1250 async global->LDS copy (ASYNCcnt) with portable fallback -----
#if __has_builtin(__builtin_amdgcn_global_load_async_to_lds_b128)
#define ASYNC_LDS 1
#endif

// The builtin expects: (int vector_size(16) AS1*, int vector_size(16) AS3*, imm, imm)
typedef int v4i_gcc __attribute__((vector_size(16)));
typedef __attribute__((address_space(1))) v4i_gcc GlobV4;
typedef __attribute__((address_space(3))) v4i_gcc LdsV4;

__device__ __forceinline__ void copy16_g2l(const void* g, void* l) {
#ifdef ASYNC_LDS
  __builtin_amdgcn_global_load_async_to_lds_b128(
      (GlobV4*)const_cast<void*>(g), (LdsV4*)l, 0, 0);
#else
  *reinterpret_cast<u32x4t*>(l) = *reinterpret_cast<const u32x4t*>(g);
#endif
}

__device__ __forceinline__ void wait_async_copies() {
#ifdef ASYNC_LDS
#if __has_builtin(__builtin_amdgcn_s_wait_asynccnt)
  __builtin_amdgcn_s_wait_asynccnt(0);
#else
  asm volatile("s_wait_asynccnt 0x0" ::: "memory");
#endif
#endif
}

// --------------------------- weight transpose ------------------------------
__global__ void k_transpose_bf16(const float* __restrict__ W,
                                 __bf16* __restrict__ Wt, int K, int N) {
  int idx = blockIdx.x * blockDim.x + threadIdx.x;
  if (idx >= K * N) return;
  int n = idx / K, k = idx % K;
  Wt[idx] = (__bf16)W[(size_t)k * N + n];
}

// ------------------------ feature GEMM (transposed out) --------------------
// OutT[j * N_NODES + i] = sum_k In[i,k] * Wt[j,k]   (j < 64, K = 128 or 64)
template <typename TIN, int K>
__global__ void k_feat_gemm_tn(const TIN* __restrict__ X,
                               const __bf16* __restrict__ Wt,
                               __bf16* __restrict__ OutT) {
  constexpr int NROW = 128, PAD = K + 8;
  __shared__ __align__(16) unsigned short sX[NROW * PAD];

  const int tid  = threadIdx.x;
  const int lane = tid & 31, wave = tid >> 5;
  const int hl = lane >> 4, l16 = lane & 15;
  const int iBase = blockIdx.x * NROW;

#pragma unroll
  for (int t = 0; t < NROW * (K / 4) / 256; ++t) {
    int u = tid + 256 * t;
    int row = u / (K / 4);
    int c   = (u % (K / 4)) * 4;
    if constexpr (sizeof(TIN) == 4) {
      f32x4 f = *reinterpret_cast<const f32x4*>(X + (size_t)(iBase + row) * K + c);
      union { __bf16 b[4]; u32x2t u2; } pk;
      pk.b[0] = (__bf16)f.x; pk.b[1] = (__bf16)f.y;
      pk.b[2] = (__bf16)f.z; pk.b[3] = (__bf16)f.w;
      *reinterpret_cast<u32x2t*>(&sX[row * PAD + c]) = pk.u2;
    } else {
      *reinterpret_cast<u32x2t*>(&sX[row * PAD + c]) =
          *reinterpret_cast<const u32x2t*>(
              reinterpret_cast<const unsigned short*>(X) + (size_t)(iBase + row) * K + c);
    }
  }
  __syncthreads();

  v8f acc[4] = {v8f{}, v8f{}, v8f{}, v8f{}};
  const int arow = wave * 16 + l16;
#pragma unroll
  for (int ks = 0; ks < K; ks += 32) {
    FragAB a;
    const unsigned short* ap = &sX[arow * PAD + ks + hl * 8];
    a.q[0] = *reinterpret_cast<const u32x4t*>(ap);
    a.q[1] = *reinterpret_cast<const u32x4t*>(ap + 16);
#pragma unroll
    for (int nt = 0; nt < 4; ++nt) {
      FragAB b;
      const __bf16* bp = Wt + (size_t)(nt * 16 + l16) * K + ks + hl * 8;
      b.q[0] = *reinterpret_cast<const u32x4t*>(bp);
      b.q[1] = *reinterpret_cast<const u32x4t*>(bp + 16);
      acc[nt] = wmma_bf16(a, b, acc[nt]);
    }
  }

#pragma unroll
  for (int nt = 0; nt < 4; ++nt) {
    int j = nt * 16 + l16;
#pragma unroll
    for (int r = 0; r < 8; ++r) {
      int i = iBase + wave * 16 + r + hl * 8;
      OutT[(size_t)j * N_NODES + i] = (__bf16)acc[nt][r];
    }
  }
}

// --------------------- dense GCN aggregation (HBM-bound) -------------------
// H[i, j] = sigmoid( sum_k A[i,k] * PT[j,k] + bias[j] )
// Double-buffered LDS; A staged via NT loads + cvt, PT via async-LDS copies.
__global__ void k_gcn_agg(const float* __restrict__ A,
                          const __bf16* __restrict__ PT,
                          const float* __restrict__ bias,
                          __bf16* __restrict__ H) {
  constexpr int NROW = 128, KC = 64, PADA = KC + 8, PADB = KC + 8;
  __shared__ __align__(16) unsigned short sA[2][NROW * PADA];  // 2 x 18.4 KB
  __shared__ __align__(16) unsigned short sP[2][H_DIM * PADB]; // 2 x  9.2 KB

  const int tid  = threadIdx.x;
  const int lane = tid & 31, wave = tid >> 5;
  const int hl = lane >> 4, l16 = lane & 15;
  const int iBase = blockIdx.x * NROW;
  const int arow  = wave * 16 + l16;

  // Per-thread staging coordinates.
  const int arow0 = tid >> 4;            // A: 8 float4 per thread per chunk
  const int acol0 = (tid & 15) * 4;
  const int pn    = tid >> 3;            // PT: 2 x 16B per thread per chunk
  const int pc    = (tid & 7) * 8;

  f32x4 areg[8];

  auto load_a = [&](int kk) {
#pragma unroll
    for (int t = 0; t < 8; ++t) {
      areg[t] = __builtin_nontemporal_load(reinterpret_cast<const f32x4*>(
          A + (size_t)(iBase + arow0 + 16 * t) * N_NODES + kk + acol0));
    }
  };
  auto store_a = [&](int buf) {
#pragma unroll
    for (int t = 0; t < 8; ++t) {
      union { __bf16 b[4]; u32x2t u2; } pk;
      pk.b[0] = (__bf16)areg[t].x; pk.b[1] = (__bf16)areg[t].y;
      pk.b[2] = (__bf16)areg[t].z; pk.b[3] = (__bf16)areg[t].w;
      *reinterpret_cast<u32x2t*>(&sA[buf][(arow0 + 16 * t) * PADA + acol0]) = pk.u2;
    }
  };
  auto stage_p = [&](int kk, int buf) {
#pragma unroll
    for (int t = 0; t < 2; ++t) {
      int n = pn + 32 * t;
      copy16_g2l(PT + (size_t)n * N_NODES + kk + pc, &sP[buf][n * PADB + pc]);
    }
  };

  v8f acc[4] = {v8f{}, v8f{}, v8f{}, v8f{}};

  // Prologue: fill buffer 0.
  load_a(0);
  stage_p(0, 0);
  store_a(0);
  wait_async_copies();
  __syncthreads();

  constexpr int ITERS = N_NODES / KC;  // 256
  for (int i = 0; i < ITERS; ++i) {
    const int cur = i & 1, nxt = cur ^ 1;
    const bool pf = (i + 1 < ITERS);
    // Kick off next chunk's loads before computing (targets the other buffer,
    // which every wave finished reading before the previous barrier).
    if (pf) {
      stage_p((i + 1) * KC, nxt);   // async -> LDS, no VGPR involvement
      load_a((i + 1) * KC);         // NT fp32 -> regs, cvt after compute
    }

#pragma unroll
    for (int ks = 0; ks < KC; ks += 32) {
      FragAB a;
      const unsigned short* ap = &sA[cur][arow * PADA + ks + hl * 8];
      a.q[0] = *reinterpret_cast<const u32x4t*>(ap);
      a.q[1] = *reinterpret_cast<const u32x4t*>(ap + 16);
#pragma unroll
      for (int nt = 0; nt < 4; ++nt) {
        FragAB b;
        const unsigned short* bp = &sP[cur][(nt * 16 + l16) * PADB + ks + hl * 8];
        b.q[0] = *reinterpret_cast<const u32x4t*>(bp);
        b.q[1] = *reinterpret_cast<const u32x4t*>(bp + 16);
        acc[nt] = wmma_bf16(a, b, acc[nt]);
      }
    }

    if (pf) {
      store_a(nxt);        // cvt + ds_store into the prefetch buffer
      wait_async_copies(); // my async PT writes have landed
    }
    __syncthreads();       // all waves' writes visible before next compute
  }

  // bias + sigmoid epilogue, bf16 row-major [N x 64]
#pragma unroll
  for (int nt = 0; nt < 4; ++nt) {
    int j = nt * 16 + l16;
    float bj = bias[j];
#pragma unroll
    for (int r = 0; r < 8; ++r) {
      int i = iBase + wave * 16 + r + hl * 8;
      float v = acc[nt][r] + bj;
      v = 1.f / (1.f + __expf(-v));
      H[(size_t)i * H_DIM + j] = (__bf16)v;
    }
  }
}

// ----------------------------- edge MLP ------------------------------------
// 128 edges/block; d = relu(ES@Wd+bd) lives in accumulators; 256->2 output
// layer fused as per-lane FMAs + 16-lane shfl_xor reduction. B fragments are
// software-pipelined (ping-pong) so loads for tile nt+1 overlap wmma of nt.
__global__ void k_edge_mlp(const int* __restrict__ edges,
                           const int* __restrict__ mask,
                           const __bf16* __restrict__ H2,
                           const __bf16* __restrict__ WdT,   // [256 x 128] bf16
                           const float* __restrict__ bd,
                           const float* __restrict__ Wo,     // [256 x 2] f32
                           const float* __restrict__ bo,
                           float* __restrict__ out) {
  constexpr int EPB = 128, KE = 128, PAD = KE + 8;
  __shared__ __align__(16) unsigned short sE[EPB * PAD];
  __shared__ float sLog[EPB][2];

  const int tid  = threadIdx.x;
  const int lane = tid & 31, wave = tid >> 5;
  const int hl = lane >> 4, l16 = lane & 15;
  const int e0 = blockIdx.x * EPB;

  // Gather concat(h2[src], h2[dst]) rows into LDS (16B units).
#pragma unroll
  for (int t = 0; t < 8; ++t) {
    int u = tid + 256 * t;
    int r = u >> 4, c = u & 15;
    int node  = edges[2 * (e0 + r) + (c >> 3)];
    int chunk = c & 7;
    *reinterpret_cast<u32x4t*>(&sE[r * PAD + c * 8]) =
        *reinterpret_cast<const u32x4t*>(H2 + (size_t)node * H_DIM + chunk * 8);
  }
  __syncthreads();

  // Preload A fragments (16 edges x K=128), reused for all 16 N-tiles.
  FragAB af[4];
  const int arow = wave * 16 + l16;
#pragma unroll
  for (int kt = 0; kt < 4; ++kt) {
    const unsigned short* ap = &sE[arow * PAD + kt * 32 + hl * 8];
    af[kt].q[0] = *reinterpret_cast<const u32x4t*>(ap);
    af[kt].q[1] = *reinterpret_cast<const u32x4t*>(ap + 16);
  }

  FragAB bb[2][4];
  auto load_b = [&](int nt, FragAB* b) {
#pragma unroll
    for (int kt = 0; kt < 4; ++kt) {
      const __bf16* bp = WdT + (size_t)(nt * 16 + l16) * KE + kt * 32 + hl * 8;
      b[kt].q[0] = *reinterpret_cast<const u32x4t*>(bp);
      b[kt].q[1] = *reinterpret_cast<const u32x4t*>(bp + 16);
    }
  };

  float ll0[8] = {0, 0, 0, 0, 0, 0, 0, 0};
  float ll1[8] = {0, 0, 0, 0, 0, 0, 0, 0};

  load_b(0, bb[0]);
#pragma unroll
  for (int nt = 0; nt < 16; ++nt) {
    if (nt + 1 < 16) load_b(nt + 1, bb[(nt + 1) & 1]);
    v8f acc = v8f{};
#pragma unroll
    for (int kt = 0; kt < 4; ++kt) acc = wmma_bf16(af[kt], bb[nt & 1][kt], acc);

    int j = nt * 16 + l16;
    float bj = bd[j], w0 = Wo[2 * j], w1 = Wo[2 * j + 1];
#pragma unroll
    for (int r = 0; r < 8; ++r) {
      float v = acc[r] + bj;
      v = v > 0.f ? v : 0.f;
      ll0[r] += v * w0;
      ll1[r] += v * w1;
    }
  }

#pragma unroll
  for (int r = 0; r < 8; ++r) {
#pragma unroll
    for (int m = 1; m < 16; m <<= 1) {
      ll0[r] += __shfl_xor(ll0[r], m, 16);
      ll1[r] += __shfl_xor(ll1[r], m, 16);
    }
    if (l16 == 0) {
      int row = wave * 16 + r + hl * 8;
      sLog[row][0] = ll0[r];
      sLog[row][1] = ll1[r];
    }
  }
  __syncthreads();

  if (tid < EPB) {
    int e = e0 + tid;
    float l0 = sLog[tid][0] + bo[0];
    float l1 = sLog[tid][1] + bo[1];
    float mx = fmaxf(l0, l1);
    float p0 = __expf(l0 - mx), p1 = __expf(l1 - mx);
    float inv = 1.f / (p0 + p1);
    float mk  = (float)mask[e];
    out[2 * e + 0] = p0 * inv * mk;
    out[2 * e + 1] = p1 * inv * mk;
  }
}

// ------------------------------- launch ------------------------------------
extern "C" void kernel_launch(void* const* d_in, const int* in_sizes, int n_in,
                              void* d_out, int out_size, void* d_ws, size_t ws_size,
                              hipStream_t stream) {
  (void)in_sizes; (void)n_in; (void)out_size; (void)ws_size;
  const float* X    = (const float*)d_in[0];
  const float* A    = (const float*)d_in[1];
  const int*   edg  = (const int*)d_in[2];
  const int*   msk  = (const int*)d_in[3];
  const float* W1   = (const float*)d_in[4];
  const float* b1   = (const float*)d_in[5];
  const float* W2   = (const float*)d_in[6];
  const float* b2   = (const float*)d_in[7];
  const float* Wd   = (const float*)d_in[8];
  const float* bd   = (const float*)d_in[9];
  const float* Wo   = (const float*)d_in[10];
  const float* bo   = (const float*)d_in[11];
  float* out = (float*)d_out;

  char* ws = (char*)d_ws;
  __bf16* W1t = (__bf16*)(ws);                               // 64 x 128
  __bf16* W2t = (__bf16*)(ws + (16 << 10));                  // 64 x 64
  __bf16* WdT = (__bf16*)(ws + (32 << 10));                  // 256 x 128
  __bf16* PT  = (__bf16*)(ws + (128 << 10));                 // 64 x 16384
  __bf16* H1  = (__bf16*)(ws + (128 << 10) + (2 << 20));     // 16384 x 64
  __bf16* H2  = (__bf16*)(ws + (128 << 10) + (4 << 20));     // 16384 x 64

  k_transpose_bf16<<<(128 * 64 + 255) / 256, 256, 0, stream>>>(W1, W1t, 128, 64);
  k_transpose_bf16<<<(64 * 64 + 255) / 256, 256, 0, stream>>>(W2, W2t, 64, 64);
  k_transpose_bf16<<<(128 * 256 + 255) / 256, 256, 0, stream>>>(Wd, WdT, 128, 256);

  k_feat_gemm_tn<float, 128><<<N_NODES / 128, 256, 0, stream>>>(X, W1t, PT);
  k_gcn_agg<<<N_NODES / 128, 256, 0, stream>>>(A, PT, b1, H1);
  k_feat_gemm_tn<__bf16, 64><<<N_NODES / 128, 256, 0, stream>>>(H1, W2t, PT);
  k_gcn_agg<<<N_NODES / 128, 256, 0, stream>>>(A, PT, b2, H2);
  k_edge_mlp<<<E_EDGES / 128, 256, 0, stream>>>(edg, msk, H2, WdT, bd, Wo, bo, out);
}
